// GAT_30030411334390
// MI455X (gfx1250) — matrix operand order repaired and checked by reading, not compiled, not run on hardware
//
#include <hip/hip_runtime.h>
#include <hip/hip_bf16.h>

#define GAT_N_NODES 50000
#define GAT_IN_FEAT 256
#define GAT_H1      512
#define GAT_H2      256
#define GAT_NCLS    40
#define GAT_NCLS_PAD 64
#define GAT_LEAKY   0.2f

typedef __attribute__((ext_vector_type(16))) __bf16 v16bf;
typedef __attribute__((ext_vector_type(8)))  __bf16 v8bf;
typedef __attribute__((ext_vector_type(8)))  float  v8f;

// ---------- elementwise f32 -> bf16 (optional fused ReLU on the input) ----------
__global__ void k_cvt_bf16(const float* __restrict__ in, __bf16* __restrict__ out,
                           long n, int relu) {
  long i = (long)blockIdx.x * blockDim.x + threadIdx.x;
  if (i < n) {
    float v = in[i];
    if (relu) v = v > 0.f ? v : 0.f;
    out[i] = (__bf16)v;
  }
}

// W [Fin, Fout] f32  ->  Wt [FoutPad, Fin] bf16 (pad rows zeroed)
__global__ void k_wt_bf16(const float* __restrict__ W, __bf16* __restrict__ Wt,
                          int Fin, int Fout, int FoutPad) {
  long i = (long)blockIdx.x * blockDim.x + threadIdx.x;
  long total = (long)FoutPad * Fin;
  if (i < total) {
    int n = (int)(i / Fin);
    int k = (int)(i % Fin);
    float v = (n < Fout) ? W[(long)k * Fout + n] : 0.f;
    Wt[i] = (__bf16)v;
  }
}

__global__ void k_init_f32(float* __restrict__ p, float v, long n) {
  long i = (long)blockIdx.x * blockDim.x + threadIdx.x;
  if (i < n) p[i] = v;
}

// ---------- WMMA GEMM: C[M,Nout] = A[M,K](bf16) * Bt[FoutPad,K](bf16)^T ----------
// One wave computes a 16x64 strip: 1 A-fragment reused across 4 B-fragments
// (4 accumulators, 4 WMMAs per 32-wide K step). 8 waves per block.
// K multiple of 32, M multiple of 16, FoutPad multiple of 64.
// A-fragment (16x32 bf16): lane L holds row L%16; K run = h*16 .. +15, h=L/16.
// B-fragment (32x16 bf16): lane L holds col L%16; K run = h*16 .. +15 (contiguous in Wt row).
__global__ void __launch_bounds__(256)
k_gemm_wmma(const __bf16* __restrict__ A, const __bf16* __restrict__ Bt,
            float* __restrict__ C, int M, int K, int Nout, int tilesNm) {
  const int lane = threadIdx.x & 31;
  const int wave = threadIdx.x >> 5;
  const long tile = (long)blockIdx.x * 8 + wave;
  const long tilesM = (long)(M >> 4);
  if (tile >= tilesM * (long)tilesNm) return;  // wave-uniform: EXEC all-1s at WMMAs
  const int tm   = (int)(tile / tilesNm);      // 16-row M tile
  const int tnm  = (int)(tile % tilesNm);      // 64-col N macro-tile
  const int half = lane >> 4;
  const int l16  = lane & 15;

  const __bf16* pa  = A  + (long)(tm * 16 + l16) * K + half * 16;
  const __bf16* pb0 = Bt + (long)(tnm * 64 + l16) * K + half * 16;
  const long bstr = 16L * K;   // elements between adjacent 16-col B tiles

  v8f acc0 = {}, acc1 = {}, acc2 = {}, acc3 = {};
  for (int kb = 0; kb < K; kb += 32) {
    v8bf alo = *(const v8bf*)(pa + kb);
    v8bf ahi = *(const v8bf*)(pa + kb + 8);
    v16bf av = __builtin_shufflevector(alo, ahi, 0,1,2,3,4,5,6,7,8,9,10,11,12,13,14,15);

    const __bf16* pb = pb0 + kb;
    v8bf b0l = *(const v8bf*)(pb);
    v8bf b0h = *(const v8bf*)(pb + 8);
    v8bf b1l = *(const v8bf*)(pb + bstr);
    v8bf b1h = *(const v8bf*)(pb + bstr + 8);
    v8bf b2l = *(const v8bf*)(pb + 2 * bstr);
    v8bf b2h = *(const v8bf*)(pb + 2 * bstr + 8);
    v8bf b3l = *(const v8bf*)(pb + 3 * bstr);
    v8bf b3h = *(const v8bf*)(pb + 3 * bstr + 8);
    v16bf b0 = __builtin_shufflevector(b0l, b0h, 0,1,2,3,4,5,6,7,8,9,10,11,12,13,14,15);
    v16bf b1 = __builtin_shufflevector(b1l, b1h, 0,1,2,3,4,5,6,7,8,9,10,11,12,13,14,15);
    v16bf b2 = __builtin_shufflevector(b2l, b2h, 0,1,2,3,4,5,6,7,8,9,10,11,12,13,14,15);
    v16bf b3 = __builtin_shufflevector(b3l, b3h, 0,1,2,3,4,5,6,7,8,9,10,11,12,13,14,15);

    acc0 = __builtin_amdgcn_wmma_f32_16x16x32_bf16(false, av, false, b0, (short)0, acc0, false, false);
    acc1 = __builtin_amdgcn_wmma_f32_16x16x32_bf16(false, av, false, b1, (short)0, acc1, false, false);
    acc2 = __builtin_amdgcn_wmma_f32_16x16x32_bf16(false, av, false, b2, (short)0, acc2, false, false);
    acc3 = __builtin_amdgcn_wmma_f32_16x16x32_bf16(false, av, false, b3, (short)0, acc3, false, false);
  }

  // C/D layout: VGPR r -> row (r + 8*half), col = lane%16
  const int rbase   = tm * 16 + half * 8;
  const int colBase = tnm * 64 + l16;
  {
    v8f* accs[4] = { &acc0, &acc1, &acc2, &acc3 };
    for (int j = 0; j < 4; ++j) {
      const int col = colBase + j * 16;
      if (col < Nout) {
        for (int r = 0; r < 8; ++r)
          C[(long)(rbase + r) * Nout + col] = (*accs[j])[r];
      }
    }
  }
}

// ---------- per-node attention dots: s = H.a_s, d = H.a_d (one wave per node) ----------
__global__ void __launch_bounds__(256)
k_attn_dots(const float* __restrict__ Hm, const float* __restrict__ as_,
            const float* __restrict__ ad_, float* __restrict__ sOut,
            float* __restrict__ dOut, int N, int F) {
  const int lane = threadIdx.x & 31;
  const long node = (long)blockIdx.x * 8 + (threadIdx.x >> 5);
  if (node >= N) return;
  const float* h = Hm + node * (long)F;
  float ss = 0.f, dd = 0.f;
  for (int f = lane; f < F; f += 32) {
    float v = h[f];
    ss += v * as_[f];
    dd += v * ad_[f];
  }
  for (int o = 16; o > 0; o >>= 1) {
    ss += __shfl_xor(ss, o, 32);
    dd += __shfl_xor(dd, o, 32);
  }
  if (lane == 0) { sOut[node] = ss; dOut[node] = dd; }
}

__device__ __forceinline__ float gat_leaky(float e) {
  return e > 0.f ? e : GAT_LEAKY * e;
}

__device__ __forceinline__ void atomicMaxF32(float* addr, float val) {
  if (val >= 0.f) atomicMax((int*)addr, __float_as_int(val));
  else            atomicMin((unsigned int*)addr, __float_as_uint(val));
}

// ---------- edge pass 1: segment max of leaky logits into m[dst] ----------
__global__ void k_edge_max(const int* __restrict__ src, const int* __restrict__ dst,
                           const float* __restrict__ s, const float* __restrict__ d,
                           float* __restrict__ m, long E) {
  long i = (long)blockIdx.x * blockDim.x + threadIdx.x;
  if (i < E) {
    int si = src[i], di = dst[i];
    atomicMaxF32(&m[di], gat_leaky(s[si] + d[di]));
  }
}

// ---------- edge pass 2: denom[dst] += exp(e - m[dst]) ----------
__global__ void k_edge_denom(const int* __restrict__ src, const int* __restrict__ dst,
                             const float* __restrict__ s, const float* __restrict__ d,
                             const float* __restrict__ m, float* __restrict__ dn, long E) {
  long i = (long)blockIdx.x * blockDim.x + threadIdx.x;
  if (i < E) {
    int si = src[i], di = dst[i];
    float e = gat_leaky(s[si] + d[di]);
    atomicAdd(&dn[di], __expf(e - m[di]));
  }
}

// ---------- edge pass 3: out[dst,:] += alpha * H[src,:]  (one wave per edge) ----------
__global__ void __launch_bounds__(256)
k_edge_agg(const int* __restrict__ src, const int* __restrict__ dst,
           const float* __restrict__ s, const float* __restrict__ d,
           const float* __restrict__ m, const float* __restrict__ dn,
           const float* __restrict__ Hm, float* __restrict__ outAgg,
           long E, int F) {
  const int lane = threadIdx.x & 31;
  const long e = (long)blockIdx.x * 8 + (threadIdx.x >> 5);
  if (e >= E) return;
  const int si = src[e], di = dst[e];
  const float lg = gat_leaky(s[si] + d[di]);
  const float alpha = __expf(lg - m[di]) / (dn[di] + 1e-9f);
  const float* hs = Hm + (long)si * F;
  float* po = outAgg + (long)di * F;
  for (int f = lane; f < F; f += 32)
    atomicAdd(po + f, alpha * hs[f]);
}

// ---------- final row softmax over 40 classes (one wave per node) ----------
__global__ void __launch_bounds__(256)
k_softmax40(const float* __restrict__ in, float* __restrict__ out, int N) {
  const int lane = threadIdx.x & 31;
  const long node = (long)blockIdx.x * 8 + (threadIdx.x >> 5);
  if (node >= N) return;
  const float* p = in + node * 40L;
  float v0 = p[lane];                               // lanes 0..31 all valid (<40)
  float v1 = (lane < 8) ? p[32 + lane] : -3.0e38f;  // lanes handle class 32+lane
  float mx = fmaxf(v0, v1);
  for (int o = 16; o > 0; o >>= 1) mx = fmaxf(mx, __shfl_xor(mx, o, 32));
  float e0 = __expf(v0 - mx);
  float e1 = (lane < 8) ? __expf(v1 - mx) : 0.f;
  float sum = e0 + e1;
  for (int o = 16; o > 0; o >>= 1) sum += __shfl_xor(sum, o, 32);
  float inv = 1.f / sum;
  out[node * 40L + lane] = e0 * inv;
  if (lane < 8) out[node * 40L + 32 + lane] = e1 * inv;
}

extern "C" void kernel_launch(void* const* d_in, const int* in_sizes, int n_in,
                              void* d_out, int out_size, void* d_ws, size_t ws_size,
                              hipStream_t stream) {
  const float* x   = (const float*)d_in[0];
  const int*   ei  = (const int*)d_in[1];
  const float* W1  = (const float*)d_in[2];
  const float* a1s = (const float*)d_in[3];
  const float* a1d = (const float*)d_in[4];
  const float* W2  = (const float*)d_in[5];
  const float* a2s = (const float*)d_in[6];
  const float* a2d = (const float*)d_in[7];
  const float* W3  = (const float*)d_in[8];
  const float* a3s = (const float*)d_in[9];
  const float* a3d = (const float*)d_in[10];
  float* out = (float*)d_out;

  const int  N = GAT_N_NODES;
  const long E = (long)in_sizes[1] / 2;
  const int* srcI = ei;
  const int* dstI = ei + E;

  // carve scratch out of d_ws (256B aligned slices)
  size_t off = 0;
  auto wsAlloc = [&](size_t bytes) -> void* {
    void* p = (char*)d_ws + off;
    off += (bytes + 255) & ~(size_t)255;
    return p;
  };
  __bf16* Xbf = (__bf16*)wsAlloc((size_t)N * 512 * sizeof(__bf16));   // activations (bf16)
  __bf16* Wt  = (__bf16*)wsAlloc((size_t)512 * 512 * sizeof(__bf16)); // W^T (bf16, padded)
  float*  Hf  = (float*) wsAlloc((size_t)N * 512 * sizeof(float));    // H = X*W (f32)
  float*  Ag  = (float*) wsAlloc((size_t)N * 512 * sizeof(float));    // aggregation out
  float*  sA  = (float*) wsAlloc((size_t)N * sizeof(float));
  float*  dA  = (float*) wsAlloc((size_t)N * sizeof(float));
  float*  mA  = (float*) wsAlloc((size_t)N * sizeof(float));
  float*  dn  = (float*) wsAlloc((size_t)N * sizeof(float));
  (void)ws_size; (void)n_in; (void)out_size;

  auto layer = [&](const float* inF32, int reluIn, const float* W,
                   const float* as_, const float* ad_,
                   int Fin, int Fout, int FoutPad) {
    const long nIn = (long)N * Fin;
    k_cvt_bf16<<<(unsigned)((nIn + 255) / 256), 256, 0, stream>>>(inF32, Xbf, nIn, reluIn);
    const long nW = (long)FoutPad * Fin;
    k_wt_bf16<<<(unsigned)((nW + 255) / 256), 256, 0, stream>>>(W, Wt, Fin, Fout, FoutPad);
    const int tilesNm = FoutPad / 64;               // 64-wide N macro-tiles per wave
    const long tiles = (long)(N / 16) * tilesNm;
    k_gemm_wmma<<<(unsigned)((tiles + 7) / 8), 256, 0, stream>>>(Xbf, Wt, Hf, N, Fin, Fout, tilesNm);
    k_attn_dots<<<(unsigned)((N + 7) / 8), 256, 0, stream>>>(Hf, as_, ad_, sA, dA, N, Fout);
    k_init_f32<<<(unsigned)((N + 255) / 256), 256, 0, stream>>>(mA, -3.0e38f, (long)N);
    k_init_f32<<<(unsigned)((N + 255) / 256), 256, 0, stream>>>(dn, 0.f, (long)N);
    const long nOut = (long)N * Fout;
    k_init_f32<<<(unsigned)((nOut + 255) / 256), 256, 0, stream>>>(Ag, 0.f, nOut);
    k_edge_max<<<(unsigned)((E + 255) / 256), 256, 0, stream>>>(srcI, dstI, sA, dA, mA, E);
    k_edge_denom<<<(unsigned)((E + 255) / 256), 256, 0, stream>>>(srcI, dstI, sA, dA, mA, dn, E);
    k_edge_agg<<<(unsigned)((E + 7) / 8), 256, 0, stream>>>(srcI, dstI, sA, dA, mA, dn, Hf, Ag, E, Fout);
  };

  // layer 1: x (no relu on input) -> Ag [N,512]
  layer(x,  0, W1, a1s, a1d, GAT_IN_FEAT, GAT_H1, GAT_H1);
  // layer 2: relu(Ag) -> Ag [N,256]   (Ag is consumed by cvt before being re-initialized)
  layer(Ag, 1, W2, a2s, a2d, GAT_H1, GAT_H2, GAT_H2);
  // layer 3: relu(Ag) -> Ag [N,40] (Fout padded to 64 for the 16x64 wave strip)
  layer(Ag, 1, W3, a3s, a3d, GAT_H2, GAT_NCLS, GAT_NCLS_PAD);
  // row softmax -> d_out
  k_softmax40<<<(unsigned)((N + 7) / 8), 256, 0, stream>>>(Ag, out, N);
}
